// TGCNet_73426760892683
// MI455X (gfx1250) — matrix operand order repaired and checked by reading, not compiled, not run on hardware
//
#include <hip/hip_runtime.h>
#include <hip/hip_bf16.h>

// ---------------------------------------------------------------------------
// TGCN forward for MI455X (gfx1250).
//   B=4, N=50000, E=800000, C_IN=C_OUT=64, OUT=32.
// H == 0 in the reference => the R gate is mathematically dead (only feeds
// H*R == 0) and the bottom halves of Lz/Lh never matter. Live pipeline:
//   deg/norm -> XW = x @ [Wz|Wh] (WMMA bf16) -> edge scatter (f32 atomics;
//   XW+AGG ~205MB, essentially L2-resident at 192MB) -> Z/Ht gates (WMMA)
//   -> head GEMM (WMMA).
// All weights are pre-packed bf16 COLUMN-MAJOR so each lane's 16 B-operand
// elements (K-contiguous, the k order this code uses for both A and B) come from two b128 loads.
// ---------------------------------------------------------------------------

#define B_    4
#define N_    50000
#define E_    800000
#define CIN_  64
#define G2_   128              // [z | h] concatenated channels
#define OUTC_ 32
#define ROWS_ (B_ * N_)        // 200000 flattened (b, n) rows
#define RT_   (ROWS_ / 16)     // 12500 row tiles of 16

typedef __attribute__((ext_vector_type(16))) __bf16         v16bf;
typedef __attribute__((ext_vector_type(8)))  float          v8f;
typedef __attribute__((ext_vector_type(8)))  unsigned short us8;

union BfVec { unsigned short us[16]; us8 h8[2]; v16bf v; };

__device__ __forceinline__ unsigned short f2bf(float f) {
  unsigned u = __float_as_uint(f);
  unsigned r = u + 0x7FFFu + ((u >> 16) & 1u);   // round-to-nearest-even
  return (unsigned short)(r >> 16);
}

__device__ __forceinline__ void pack4(BfVec& a, int o, float4 p) {
  a.us[o + 0] = f2bf(p.x); a.us[o + 1] = f2bf(p.y);
  a.us[o + 2] = f2bf(p.z); a.us[o + 3] = f2bf(p.w);
}

// A operand (16x32 bf16): lane&15 = row; lane>>4 selects K-phase.
// Elements 0..15 hold K = kb..kb+15,
// kb = 16*(lane>>4). The run is 32B-contiguous in a row-major source.
// B operand (32x16 bf16): lane&15 = column; elements 0..15 hold
// K = 16*(lane>>4) + 0..15 -> 32B-contiguous in a column-major source.

// ---------------------------------------------------------------------------
// Weight prep: fp32 -> bf16 column-major operand buffers (tiny, once/launch).
//   WcatT[c][k] : c<64 -> Wz[k][c], else Wh[k][c-64]     (128 cols x 64 K)
//   LtT[g][o][k]: g=0 -> Lz top half, g=1 -> Lh top half (2 x 64 x 64)
//   HwbT[o][k]  : Hw[k][o]                               (32 x 64)
// ---------------------------------------------------------------------------
__global__ void tgcn_prep_weights(const float* __restrict__ Wz,
                                  const float* __restrict__ Wh,
                                  const float* __restrict__ Lz,
                                  const float* __restrict__ Lh,
                                  const float* __restrict__ Hw,
                                  unsigned short* __restrict__ WcatT,
                                  unsigned short* __restrict__ LtT,
                                  unsigned short* __restrict__ HwbT) {
  int i = blockIdx.x * blockDim.x + threadIdx.x;
  if (i < G2_ * 64) {
    int c = i / 64, k = i % 64;
    float v = (c < 64) ? Wz[k * 64 + c] : Wh[k * 64 + (c - 64)];
    WcatT[i] = f2bf(v);
  }
  if (i < 2 * 64 * 64) {
    int g = i / 4096, r = i % 4096;          // r = o*64 + k
    int o = r / 64, k = r % 64;
    const float* L = (g == 0) ? Lz : Lh;     // top halves: row k < 64
    LtT[i] = f2bf(L[k * 64 + o]);
  }
  if (i < OUTC_ * 64) {
    int o = i / 64, k = i % 64;
    HwbT[i] = f2bf(Hw[k * OUTC_ + o]);
  }
}

// ---------------------------------------------------------------------------
// Degree / norm
// ---------------------------------------------------------------------------
__global__ void tgcn_deg_init(float* __restrict__ deg) {
  int i = blockIdx.x * blockDim.x + threadIdx.x;
  if (i < N_) deg[i] = 1.0f;                 // self-loop contribution
}

__global__ void tgcn_deg_scatter(const int* __restrict__ ei,
                                 const float* __restrict__ w,
                                 float* __restrict__ deg) {
  int e = blockIdx.x * blockDim.x + threadIdx.x;
  if (e < E_) atomicAdd(&deg[ei[E_ + e]], w[e]);
}

__global__ void tgcn_norm(const int* __restrict__ ei,
                          const float* __restrict__ w,
                          const float* __restrict__ deg,
                          float* __restrict__ nrm) {
  int e = blockIdx.x * blockDim.x + threadIdx.x;
  if (e < E_) nrm[e] = __frsqrt_rn(deg[ei[e]]) * w[e] * __frsqrt_rn(deg[ei[E_ + e]]);
}

// ---------------------------------------------------------------------------
// XW = x @ Wcat  (ROWS x 64)*(64 x 128) f32 acc, 2x v_wmma_f32_16x16x32_bf16.
// grid (RT_, 2), block 128 (4 waves); wave -> one 16x16 output tile.
// ---------------------------------------------------------------------------
__global__ void tgcn_gemm_xw(const float* __restrict__ x,
                             const unsigned short* __restrict__ WcatT,
                             float* __restrict__ XW) {
  int lane = threadIdx.x & 31, wave = threadIdx.x >> 5;
  int rowbase = blockIdx.x * 16;
  int colTile = blockIdx.y * 4 + wave;       // 0..7
  int col  = colTile * 16 + (lane & 15);
  int rsel = lane >> 4;
  const float4* xr4 = (const float4*)(x + (size_t)(rowbase + (lane & 15)) * CIN_);
  const unsigned short* wc = WcatT + (size_t)col * 64;

  v8f c = {0.f, 0.f, 0.f, 0.f, 0.f, 0.f, 0.f, 0.f};
#pragma unroll
  for (int kc = 0; kc < 64; kc += 32) {
    int kb = kc + rsel * 16;
    BfVec a, b;
    pack4(a, 0,  xr4[kb / 4]);     pack4(a, 4,  xr4[kb / 4 + 1]);
    pack4(a, 8,  xr4[kb / 4 + 2]); pack4(a, 12, xr4[kb / 4 + 3]);
    const us8* wp = (const us8*)(wc + kc + rsel * 16);
    b.h8[0] = wp[0]; b.h8[1] = wp[1];
    c = __builtin_amdgcn_wmma_f32_16x16x32_bf16(false, a.v, false, b.v,
                                                (short)0, c, false, false);
  }
#pragma unroll
  for (int i = 0; i < 8; ++i)
    XW[(size_t)(rowbase + rsel * 8 + i) * G2_ + col] = c[i];
}

// AGG init = self-loop term: XW[row] * (1/deg[n])   (1/deg == dinv*1*dinv)
__global__ void tgcn_agg_init(const float4* __restrict__ XW4,
                              const float* __restrict__ deg,
                              float4* __restrict__ AGG4) {
  int i = blockIdx.x * blockDim.x + threadIdx.x;   // ROWS_*32 float4s
  int n = (i >> 5) % N_;                           // row = i / (G2_/4)
  float s = 1.0f / deg[n];
  float4 v = XW4[i];
  v.x *= s; v.y *= s; v.z *= s; v.w *= s;
  AGG4[i] = v;
}

// Edge scatter: AGG[b,tgt,:] += norm_e * XW[b,src,:]. One wave per edge;
// each lane owns one float4 (128 channels / 32 lanes). AGG stays in the
// 192MB L2, so these atomics are on-chip. block 256 = 8 edges.
__global__ void tgcn_edge_scatter(const int* __restrict__ ei,
                                  const float* __restrict__ nrm,
                                  const float* __restrict__ XW,
                                  float* __restrict__ AGG) {
  int e = blockIdx.x * 8 + (threadIdx.x >> 5);
  if (e >= E_) return;
  int lane = threadIdx.x & 31;
  int s = ei[e], t = ei[E_ + e];
  float nm = nrm[e];
#pragma unroll
  for (int b = 0; b < B_; ++b) {
    const float* src = XW  + (size_t)(b * N_ + s) * G2_;
    float*       dst = AGG + (size_t)(b * N_ + t) * G2_;
    if (b + 1 < B_)                           // gfx1250 global_prefetch_b8
      __builtin_prefetch(XW + (size_t)((b + 1) * N_ + s) * G2_ + lane * 4, 0, 1);
    float4 v = ((const float4*)src)[lane];
    atomicAdd(&dst[lane * 4 + 0], nm * v.x);
    atomicAdd(&dst[lane * 4 + 1], nm * v.y);
    atomicAdd(&dst[lane * 4 + 2], nm * v.z);
    atomicAdd(&dst[lane * 4 + 3], nm * v.w);
  }
}

// ---------------------------------------------------------------------------
// Gates: Z = sigmoid((AGGz+bz) @ Lz_top + Lzb), Ht = tanh((AGGh+bh) @ Lh_top
// + Lhb), HR = relu((1-Z)*Ht) stored bf16 for the head GEMM.
// grid (RT_), block 128; wave = 16-col tile (4 tiles of 64 cols).
// ---------------------------------------------------------------------------
__global__ void tgcn_gates(const float* __restrict__ AGG,
                           const unsigned short* __restrict__ LtT,
                           const float* __restrict__ bz,
                           const float* __restrict__ bh,
                           const float* __restrict__ Lzb,
                           const float* __restrict__ Lhb,
                           unsigned short* __restrict__ HR) {
  int lane = threadIdx.x & 31, wave = threadIdx.x >> 5;
  int rowbase = blockIdx.x * 16;
  int col  = wave * 16 + (lane & 15);
  int rsel = lane >> 4;
  const float4* ag4 = (const float4*)(AGG + (size_t)(rowbase + (lane & 15)) * G2_);
  const float4* bz4 = (const float4*)bz;
  const float4* bh4 = (const float4*)bh;
  const unsigned short* lz = LtT + (size_t)col * 64;          // gate 0 col
  const unsigned short* lh = LtT + 4096 + (size_t)col * 64;   // gate 1 col

  v8f zc, hc;
  float zb0 = Lzb[col], hb0 = Lhb[col];
#pragma unroll
  for (int i = 0; i < 8; ++i) { zc[i] = zb0; hc[i] = hb0; }

#pragma unroll
  for (int kc = 0; kc < 64; kc += 32) {
    int kb = kc + rsel * 16;                  // multiple of 4
    BfVec az, ah, bzv, bhv;
    float4 p, q;
#pragma unroll
    for (int g = 0; g < 4; ++g) {             // 4 float4 chunks of A per gate
      int kk = kb + g * 4;                           // kb, kb+4, kb+8, kb+12
      p = ag4[kk / 4];            q = bz4[kk / 4];
      p.x += q.x; p.y += q.y; p.z += q.z; p.w += q.w;
      pack4(az, g * 4, p);
      p = ag4[16 + kk / 4];       q = bh4[kk / 4];   // h channels at +64
      p.x += q.x; p.y += q.y; p.z += q.z; p.w += q.w;
      pack4(ah, g * 4, p);
    }
    const us8* zp = (const us8*)(lz + kc + rsel * 16);
    const us8* hp = (const us8*)(lh + kc + rsel * 16);
    bzv.h8[0] = zp[0]; bzv.h8[1] = zp[1];
    bhv.h8[0] = hp[0]; bhv.h8[1] = hp[1];
    zc = __builtin_amdgcn_wmma_f32_16x16x32_bf16(false, az.v, false, bzv.v,
                                                 (short)0, zc, false, false);
    hc = __builtin_amdgcn_wmma_f32_16x16x32_bf16(false, ah.v, false, bhv.v,
                                                 (short)0, hc, false, false);
  }
#pragma unroll
  for (int i = 0; i < 8; ++i) {
    float z  = 1.0f / (1.0f + __expf(-zc[i]));
    float ht = tanhf(hc[i]);
    float hn = (1.0f - z) * ht;               // Hn = Z*H + (1-Z)*Ht, H==0
    hn = fmaxf(hn, 0.0f);                     // relu
    HR[(size_t)(rowbase + rsel * 8 + i) * 64 + col] = f2bf(hn);
  }
}

// ---------------------------------------------------------------------------
// Head: out = HR @ Hw + Hb  ((ROWS x 64)*(64 x 32)).
// grid (RT_), block 64; wave = one of two 16-col tiles. A is already bf16
// row-major -> two b128 loads per tile per lane.
// ---------------------------------------------------------------------------
__global__ void tgcn_head(const unsigned short* __restrict__ HR,
                          const unsigned short* __restrict__ HwbT,
                          const float* __restrict__ Hb,
                          float* __restrict__ out) {
  int lane = threadIdx.x & 31, wave = threadIdx.x >> 5;
  int rowbase = blockIdx.x * 16;
  int col  = wave * 16 + (lane & 15);
  int rsel = lane >> 4;
  const unsigned short* hr = HR + (size_t)(rowbase + (lane & 15)) * 64;
  const unsigned short* hw = HwbT + (size_t)col * 64;

  v8f c;
  float hb = Hb[col];
#pragma unroll
  for (int i = 0; i < 8; ++i) c[i] = hb;

#pragma unroll
  for (int kc = 0; kc < 64; kc += 32) {
    int kb = kc + rsel * 16;
    BfVec a, b;
    a.h8[0] = *(const us8*)(hr + kb);
    a.h8[1] = *(const us8*)(hr + kb + 8);
    const us8* wp = (const us8*)(hw + kc + rsel * 16);
    b.h8[0] = wp[0]; b.h8[1] = wp[1];
    c = __builtin_amdgcn_wmma_f32_16x16x32_bf16(false, a.v, false, b.v,
                                                (short)0, c, false, false);
  }
#pragma unroll
  for (int i = 0; i < 8; ++i)
    out[(size_t)(rowbase + rsel * 8 + i) * OUTC_ + col] = c[i];
}

// ---------------------------------------------------------------------------
// kernel_launch
// ---------------------------------------------------------------------------
extern "C" void kernel_launch(void* const* d_in, const int* in_sizes, int n_in,
                              void* d_out, int out_size, void* d_ws, size_t ws_size,
                              hipStream_t stream) {
  const float* x   = (const float*)d_in[0];
  const int*   ei  = (const int*)  d_in[1];   // [2, E] (src row, tgt row)
  const float* ew  = (const float*)d_in[2];
  const float* Wz  = (const float*)d_in[3];
  const float* bz  = (const float*)d_in[4];
  // d_in[5] = Wr, d_in[6] = br : dead (R only feeds H*R with H == 0)
  const float* Wh  = (const float*)d_in[7];
  const float* bh  = (const float*)d_in[8];
  const float* Lz  = (const float*)d_in[9];
  const float* Lzb = (const float*)d_in[10];
  // d_in[11] = Lr, d_in[12] = Lrb : dead
  const float* Lh  = (const float*)d_in[13];
  const float* Lhb = (const float*)d_in[14];
  const float* Hw  = (const float*)d_in[15];
  const float* Hb  = (const float*)d_in[16];
  float* out = (float*)d_out;

  // Workspace carve (~234 MB total, 16B-aligned slabs).
  float* XW  = (float*)d_ws;                           // ROWS_*128 f32
  float* AGG = XW + (size_t)ROWS_ * G2_;               // ROWS_*128 f32
  unsigned short* HR = (unsigned short*)(AGG + (size_t)ROWS_ * G2_); // ROWS_*64
  float* nrm = (float*)(HR + (size_t)ROWS_ * 64);      // E_ f32
  float* deg = nrm + E_;                               // N_ f32
  unsigned short* WcatT = (unsigned short*)(deg + N_); // 128*64 bf16 (col-major)
  unsigned short* LtT   = WcatT + G2_ * 64;            // 2*64*64 bf16 (col-major)
  unsigned short* HwbT  = LtT + 2 * 64 * 64;           // 32*64 bf16 (col-major)

  tgcn_prep_weights<<<(8192 + 255) / 256, 256, 0, stream>>>(
      Wz, Wh, Lz, Lh, Hw, WcatT, LtT, HwbT);
  tgcn_deg_init<<<(N_ + 255) / 256, 256, 0, stream>>>(deg);
  tgcn_deg_scatter<<<(E_ + 255) / 256, 256, 0, stream>>>(ei, ew, deg);
  tgcn_norm<<<(E_ + 255) / 256, 256, 0, stream>>>(ei, ew, deg, nrm);

  tgcn_gemm_xw<<<dim3(RT_, 2), 128, 0, stream>>>(x, WcatT, XW);
  tgcn_agg_init<<<(ROWS_ * (G2_ / 4)) / 256, 256, 0, stream>>>(
      (const float4*)XW, deg, (float4*)AGG);
  tgcn_edge_scatter<<<(E_ + 7) / 8, 256, 0, stream>>>(ei, nrm, XW, AGG);

  tgcn_gates<<<RT_, 128, 0, stream>>>(AGG, LtT, bz, bh, Lzb, Lhb, HR);
  tgcn_head<<<RT_, 64, 0, stream>>>(HR, HwbT, Hb, out);
}
